// CausalSelfAttention_27178553049708
// MI455X (gfx1250) — matrix operand order repaired and checked by reading, not compiled, not run on hardware
//
#include <hip/hip_runtime.h>
#include <hip/hip_bf16.h>

// ---------------- types ----------------
typedef __attribute__((ext_vector_type(16))) __bf16 v16bf;
typedef __attribute__((ext_vector_type(8)))  __bf16 v8bf;
typedef __attribute__((ext_vector_type(8)))  float  v8f;

#define D_MODEL 1024
#define NH      16
#define DH      64
#define SEQ_T   2048
#define BATCH   4
#define M_TOT   (BATCH * SEQ_T)   // 8192 tokens

// ---------------- WMMA helpers ----------------
__device__ __forceinline__ v8f wmma_bf16(v16bf a, v16bf b, v8f c) {
  // D = A(16x32 bf16) x B(32x16 bf16) + C(16x16 f32)
  return __builtin_amdgcn_wmma_f32_16x16x32_bf16(
      /*neg_a=*/false, a, /*neg_b=*/false, b,
      /*c_mod=*/(short)0, c, /*reuse_a=*/false, /*reuse_b=*/false);
}

// A-matrix fragment (16x32, bf16). Memory: row-major, K contiguous, stride ld.
// Lane l: row = l&15, half = l>>4. Element e -> K = k0 + e + 8*half + (e>=8 ? 8 : 0).
__device__ __forceinline__ v16bf load_a_frag(const __bf16* base, int ld,
                                             int row0, int k0, int lane) {
  int m = lane & 15, half = lane >> 4;
  const __bf16* p = base + (size_t)(row0 + m) * ld + k0 + half * 8;
  v8bf lo = *(const v8bf*)(p);        // K = k0 + 8h + 0..7
  v8bf hi = *(const v8bf*)(p + 16);   // K = k0 + 8h + 16..23
  v16bf r;
#pragma unroll
  for (int i = 0; i < 8; ++i) { r[i] = lo[i]; r[i + 8] = hi[i]; }
  return r;
}

// B-matrix fragment (32x16, bf16). Memory: for each output column n, K is
// contiguous with stride ld between columns (i.e. "column-major B" storage).
// Lane l: col = l&15, half = l>>4. Element e -> K = k0 + e + 8*half + (e>=8 ? 8 : 0).
__device__ __forceinline__ v16bf load_b_frag(const __bf16* base, int ld,
                                             int col0, int k0, int lane) {
  int n = lane & 15, half = lane >> 4;
  const __bf16* p = base + (size_t)(col0 + n) * ld + k0 + half * 8;
  v8bf lo = *(const v8bf*)(p);
  v8bf hi = *(const v8bf*)(p + 16);
  v16bf r;
#pragma unroll
  for (int i = 0; i < 8; ++i) { r[i] = lo[i]; r[i + 8] = hi[i]; }
  return r;
}

// ---------------- CDNA5 async copy: global -> LDS (ASYNCcnt tracked) ----------
// 256 threads stage a 64(row) x 32(k) bf16 tile (4 KB): one b128 chunk each.
// wt is (N, K=1024) row-major bf16, so k is contiguous within a row.
__device__ __forceinline__ void async_stage_b(const __bf16* wt, int n0, int k0,
                                              __bf16* buf, int tid) {
  int nr = tid >> 2;   // 0..63 : row within tile
  int kc = tid & 3;    // 0..3  : 8-element (16 B) chunk within 32-k slice
  const __bf16* g = wt + (size_t)(n0 + nr) * D_MODEL + k0 + kc * 8;
  unsigned lds_off = (unsigned)(size_t)(buf + nr * 32 + kc * 8);
  unsigned long long ga = (unsigned long long)(size_t)g;
  asm volatile("global_load_async_to_lds_b128 %0, %1, off"
               :: "v"(lds_off), "v"(ga) : "memory");
}
#define ASYNC_WAIT() asm volatile("s_wait_asynccnt 0" ::: "memory")

// ---------------- conversion / transpose kernels ----------------
__global__ void __launch_bounds__(256)
cvt_f32_to_bf16(const float* __restrict__ in, __bf16* __restrict__ out) {
  size_t i = (size_t)blockIdx.x * 256 + threadIdx.x;
  out[i] = (__bf16)in[i];
}

// in: (K=1024, N) row-major f32  ->  out: (N, K) row-major bf16
__global__ void __launch_bounds__(256)
transpose_to_bf16(const float* __restrict__ in, __bf16* __restrict__ out, int N) {
  size_t i = (size_t)blockIdx.x * 256 + threadIdx.x;   // i = k*N + n
  int k = (int)(i / N);
  int n = (int)(i % N);
  out[(size_t)n * D_MODEL + k] = (__bf16)in[i];
}

// ---------------- QKV GEMM: (8192 x 1024) x (1024 x 3072) ----------------
// Block: 8 waves x 16 rows = 128 rows, shared 64-col B tile staged async->LDS
// (double buffered). Epilogue scatters Q,K (B,H,T,Dh) and V^T (B,H,Dh,T).
__global__ void __launch_bounds__(256)
gemm_qkv_kernel(const __bf16* __restrict__ xb, const __bf16* __restrict__ wt,
                const float* __restrict__ bias,
                __bf16* __restrict__ Qb, __bf16* __restrict__ Kb,
                __bf16* __restrict__ Vtb) {
  __shared__ __bf16 bufB[2][64 * 32];
  int tid  = threadIdx.x;
  int lane = tid & 31;
  int wv   = tid >> 5;
  int n0   = blockIdx.x * 64;                 // 48 col blocks
  int row0 = blockIdx.y * 128 + wv * 16;      // 64 row blocks

  async_stage_b(wt, n0, 0, &bufB[0][0], tid);
  v16bf a_cur = load_a_frag(xb, D_MODEL, row0, 0, lane);
  ASYNC_WAIT();
  __syncthreads();

  v8f acc[4] = {v8f{}, v8f{}, v8f{}, v8f{}};
  int cur = 0;
  for (int k0 = 0; k0 < D_MODEL; k0 += 32) {
    int nxt = k0 + 32;
    v16bf a_next = a_cur;
    if (nxt < D_MODEL) {                       // prefetch next step
      async_stage_b(wt, n0, nxt, &bufB[cur ^ 1][0], tid);
      a_next = load_a_frag(xb, D_MODEL, row0, nxt, lane);
    }
#pragma unroll
    for (int s = 0; s < 4; ++s) {
      v16bf b = load_b_frag(&bufB[cur][0], 32, s * 16, 0, lane);
      acc[s] = wmma_bf16(a_cur, b, acc[s]);
    }
    if (nxt < D_MODEL) ASYNC_WAIT();
    __syncthreads();
    a_cur = a_next;
    cur ^= 1;
  }

  int n = lane & 15, half = lane >> 4;
#pragma unroll
  for (int s = 0; s < 4; ++s) {
    int ncol  = n0 + s * 16 + n;
    int which = ncol >> 10;        // 0=q 1=k 2=v
    int c     = ncol & 1023;
    int h     = c >> 6, d = c & 63;
    float bv  = bias[ncol];
#pragma unroll
    for (int r = 0; r < 8; ++r) {
      int t  = row0 + r + half * 8;          // global token
      int b_ = t >> 11, tt = t & (SEQ_T - 1);
      __bf16 o = (__bf16)(acc[s][r] + bv);
      size_t bh = (size_t)(b_ * NH + h);
      if (which == 0)      Qb[(bh * SEQ_T + tt) * DH + d]  = o;
      else if (which == 1) Kb[(bh * SEQ_T + tt) * DH + d]  = o;
      else                 Vtb[(bh * DH + d) * SEQ_T + tt] = o;   // transposed
    }
  }
}

// ---------------- Flash attention: one wave = 16 queries ----------------
__global__ void __launch_bounds__(256)
attn_kernel(const __bf16* __restrict__ Qb, const __bf16* __restrict__ Kb,
            const __bf16* __restrict__ Vtb, __bf16* __restrict__ attn_out) {
  __shared__ __bf16 ldsP[8][16 * 32];        // per-wave 16x32 bf16 P tile

  int lane = threadIdx.x & 31;
  int widx = threadIdx.x >> 5;
  int wid  = blockIdx.x * 8 + widx;
  int qt   = wid & 127;        // 128 query tiles
  int bh   = wid >> 7;         // 0..63  (b*16 + h)
  int q0   = qt * 16;

  const __bf16* Qp = Qb  + (size_t)bh * SEQ_T * DH;
  const __bf16* Kp = Kb  + (size_t)bh * SEQ_T * DH;
  const __bf16* Vp = Vtb + (size_t)bh * DH * SEQ_T;

  v16bf qa0 = load_a_frag(Qp, DH, q0, 0,  lane);
  v16bf qa1 = load_a_frag(Qp, DH, q0, 32, lane);

  v8f o[4] = {v8f{}, v8f{}, v8f{}, v8f{}};
  float mi[8], li[8];
#pragma unroll
  for (int r = 0; r < 8; ++r) { mi[r] = -1e30f; li[r] = 0.0f; }

  int n = lane & 15, half = lane >> 4;
  __bf16* myP = &ldsP[widx][0];

  for (int j0 = 0; j0 <= q0 + 15; j0 += 32) {
    bool do2 = (j0 + 16) <= (q0 + 15);
    v8f s0 = v8f{}, s1 = v8f{};
    {
      v16bf kb = load_b_frag(Kp, DH, j0, 0, lane);
      s0 = wmma_bf16(qa0, kb, s0);
      kb = load_b_frag(Kp, DH, j0, 32, lane);
      s0 = wmma_bf16(qa1, kb, s0);
    }
    if (do2) {
      v16bf kb = load_b_frag(Kp, DH, j0 + 16, 0, lane);
      s1 = wmma_bf16(qa0, kb, s1);
      kb = load_b_frag(Kp, DH, j0 + 16, 32, lane);
      s1 = wmma_bf16(qa1, kb, s1);
    }

#pragma unroll
    for (int r = 0; r < 8; ++r) {
      int q = q0 + r + half * 8;
      float a = ((j0 + n) <= q)              ? s0[r] * 0.125f : -1e30f;
      float b = (do2 && (j0 + 16 + n) <= q)  ? s1[r] * 0.125f : -1e30f;
      // row max across the 16-lane half (wave32: masks < 16 stay in half)
      float mstep = fmaxf(a, b);
#pragma unroll
      for (int w = 1; w < 16; w <<= 1) mstep = fmaxf(mstep, __shfl_xor(mstep, w, 32));
      float mnew  = fmaxf(mi[r], mstep);
      float alpha = __expf(mi[r] - mnew);
      float p0 = __expf(a - mnew);
      float p1 = __expf(b - mnew);
      float rsum = p0 + p1;
#pragma unroll
      for (int w = 1; w < 16; w <<= 1) rsum += __shfl_xor(rsum, w, 32);
      li[r] = li[r] * alpha + rsum;
      mi[r] = mnew;
#pragma unroll
      for (int dt = 0; dt < 4; ++dt) o[dt][r] *= alpha;
      int m = r + half * 8;                         // C-layout row
      myP[m * 32 + n]      = (__bf16)p0;            // stage P in LDS
      myP[m * 32 + 16 + n] = (__bf16)p1;
    }

    // P (16x32) back as A-fragment (ds_load), V^T columns contiguous in key
    v16bf pa = load_a_frag(myP, 32, 0, 0, lane);
#pragma unroll
    for (int dt = 0; dt < 4; ++dt) {
      v16bf vb = load_b_frag(Vp, SEQ_T, dt * 16, j0, lane);
      o[dt] = wmma_bf16(pa, vb, o[dt]);
    }
  }

  // epilogue: normalize and write (B,T,C) bf16
  int b_ = bh >> 4, h = bh & 15;
#pragma unroll
  for (int dt = 0; dt < 4; ++dt) {
#pragma unroll
    for (int r = 0; r < 8; ++r) {
      int t   = q0 + r + half * 8;
      int col = h * DH + dt * 16 + n;
      attn_out[((size_t)(b_ * SEQ_T + t)) * D_MODEL + col] =
          (__bf16)(o[dt][r] / li[r]);
    }
  }
}

// ---------------- Output projection: (8192 x 1024) x (1024 x 1024) ----------------
__global__ void __launch_bounds__(256)
gemm_out_kernel(const __bf16* __restrict__ ab, const __bf16* __restrict__ wt,
                const float* __restrict__ bias, float* __restrict__ out) {
  __shared__ __bf16 bufB[2][64 * 32];
  int tid  = threadIdx.x;
  int lane = tid & 31;
  int wv   = tid >> 5;
  int n0   = blockIdx.x * 64;                 // 16 col blocks
  int row0 = blockIdx.y * 128 + wv * 16;      // 64 row blocks

  async_stage_b(wt, n0, 0, &bufB[0][0], tid);
  v16bf a_cur = load_a_frag(ab, D_MODEL, row0, 0, lane);
  ASYNC_WAIT();
  __syncthreads();

  v8f acc[4] = {v8f{}, v8f{}, v8f{}, v8f{}};
  int cur = 0;
  for (int k0 = 0; k0 < D_MODEL; k0 += 32) {
    int nxt = k0 + 32;
    v16bf a_next = a_cur;
    if (nxt < D_MODEL) {
      async_stage_b(wt, n0, nxt, &bufB[cur ^ 1][0], tid);
      a_next = load_a_frag(ab, D_MODEL, row0, nxt, lane);
    }
#pragma unroll
    for (int s = 0; s < 4; ++s) {
      v16bf b = load_b_frag(&bufB[cur][0], 32, s * 16, 0, lane);
      acc[s] = wmma_bf16(a_cur, b, acc[s]);
    }
    if (nxt < D_MODEL) ASYNC_WAIT();
    __syncthreads();
    a_cur = a_next;
    cur ^= 1;
  }

  int n = lane & 15, half = lane >> 4;
#pragma unroll
  for (int s = 0; s < 4; ++s) {
    int ncol = n0 + s * 16 + n;
    float bv = bias[ncol];
#pragma unroll
    for (int r = 0; r < 8; ++r) {
      int t = row0 + r + half * 8;
      out[(size_t)t * D_MODEL + ncol] = acc[s][r] + bv;
    }
  }
}

// ---------------- launcher ----------------
extern "C" void kernel_launch(void* const* d_in, const int* in_sizes, int n_in,
                              void* d_out, int out_size, void* d_ws, size_t ws_size,
                              hipStream_t stream) {
  const float* x     = (const float*)d_in[0];   // (4,2048,1024)
  const float* Wqkv  = (const float*)d_in[1];   // (1024,3072)
  const float* bqkv  = (const float*)d_in[2];   // (3072,)
  const float* Wout  = (const float*)d_in[3];   // (1024,1024)
  const float* bout  = (const float*)d_in[4];   // (1024,)
  float* out = (float*)d_out;

  char* ws = (char*)d_ws;
  __bf16* xb    = (__bf16*)ws;  ws += (size_t)M_TOT * D_MODEL * 2;       // 16 MB
  __bf16* wqt   = (__bf16*)ws;  ws += (size_t)3 * D_MODEL * D_MODEL * 2; //  6 MB
  __bf16* wot   = (__bf16*)ws;  ws += (size_t)D_MODEL * D_MODEL * 2;     //  2 MB
  __bf16* Qb    = (__bf16*)ws;  ws += (size_t)M_TOT * D_MODEL * 2;       // 16 MB
  __bf16* Kb    = (__bf16*)ws;  ws += (size_t)M_TOT * D_MODEL * 2;       // 16 MB
  __bf16* Vtb   = (__bf16*)ws;  ws += (size_t)M_TOT * D_MODEL * 2;       // 16 MB
  __bf16* attnb = (__bf16*)ws;  ws += (size_t)M_TOT * D_MODEL * 2;       // 16 MB

  // 1) fp32 -> bf16 conversions (+ weight transposes for contiguous B-frags)
  cvt_f32_to_bf16<<<(M_TOT * D_MODEL) / 256, 256, 0, stream>>>(x, xb);
  transpose_to_bf16<<<(D_MODEL * 3 * D_MODEL) / 256, 256, 0, stream>>>(Wqkv, wqt, 3 * D_MODEL);
  transpose_to_bf16<<<(D_MODEL * D_MODEL) / 256, 256, 0, stream>>>(Wout, wot, D_MODEL);

  // 2) QKV projection: 48 col-blocks x 64 row-blocks, 8 waves/block
  gemm_qkv_kernel<<<dim3(48, 64), 256, 0, stream>>>(xb, wqt, bqkv, Qb, Kb, Vtb);

  // 3) flash attention: 64 (b,h) x 128 q-tiles, 8 waves/block
  attn_kernel<<<(64 * 128) / 8, 256, 0, stream>>>(Qb, Kb, Vtb, attnb);

  // 4) output projection: 16 col-blocks x 64 row-blocks
  gemm_out_kernel<<<dim3(16, 64), 256, 0, stream>>>(attnb, wot, bout, out);
}
